// GaussianQueryModule_16037407883748
// MI455X (gfx1250) — matrix operand order repaired and checked by reading, not compiled, not run on hardware
//
#include <hip/hip_runtime.h>
#include <hip/hip_bf16.h>
#include <math.h>

typedef __bf16 bf16_t;
typedef __attribute__((ext_vector_type(16))) __bf16 v16bf;
typedef __attribute__((ext_vector_type(8)))  __bf16 v8bf;
typedef __attribute__((ext_vector_type(8)))  float  v8f;

#define NB    4
#define CIN   64
#define HIN   112
#define WIN   112
#define NG_   100
#define PP    49
#define HO    224
#define WO    224
#define HIDN  256
#define KCONV 576        // 64*9 im2col depth
#define ROWS2 512        // coef(256)+freq(256) fused output rows
#define QPI   50176      // queries per image (224*224)
#define QTOT  200704     // 4*QPI
#define PIXIN 12544      // 112*112
#define PIXO  50176      // 224*224

__device__ __forceinline__ bf16_t f2bf(float f) { return (bf16_t)f; }

// ---------------------------------------------------------------- conv1 3x3 64->64 + relu (f32)
__global__ void k_conv1(const float* __restrict__ feat, const float* __restrict__ w1,
                        const float* __restrict__ b1, float* __restrict__ hout) {
  __shared__ float tile[8][18][18];
  const int t = threadIdx.x;
  const int tyi = blockIdx.x / 7, txi = blockIdx.x % 7;
  const int co = blockIdx.y, b = blockIdx.z;
  const int ty = t >> 4, tx = t & 15;
  const int y0 = tyi * 16, x0 = txi * 16;
  float acc = b1[co];
  for (int ci0 = 0; ci0 < CIN; ci0 += 8) {
    for (int e = t; e < 8 * 18 * 18; e += 256) {
      int ci8 = e / 324, r = e % 324, sy = r / 18, sx = r % 18;
      int gy = y0 - 1 + sy, gx = x0 - 1 + sx;
      float v = 0.f;
      if (gy >= 0 && gy < HIN && gx >= 0 && gx < WIN)
        v = feat[((b * CIN + ci0 + ci8) * HIN + gy) * WIN + gx];
      tile[ci8][sy][sx] = v;
    }
    __syncthreads();
#pragma unroll
    for (int ci8 = 0; ci8 < 8; ++ci8)
#pragma unroll
      for (int ky = 0; ky < 3; ++ky)
#pragma unroll
        for (int kx = 0; kx < 3; ++kx)
          acc += tile[ci8][ty + ky][tx + kx] *
                 w1[((co * CIN + ci0 + ci8) * 3 + ky) * 3 + kx];
    __syncthreads();
  }
  hout[((b * CIN + co) * HIN + y0 + ty) * WIN + x0 + tx] = fmaxf(acc, 0.f);
}

// ------------------- fused 1x1 conv -> softmax -> weighted sums; deterministic per-block partials
__global__ void k_logits(const float* __restrict__ hin, const float* __restrict__ w2,
                         const float* __restrict__ b2,
                         const float* __restrict__ sigx, const float* __restrict__ sigy,
                         const float* __restrict__ opac, const float* __restrict__ rho_,
                         float* __restrict__ partial) {
  __shared__ float w2s[NG_ * CIN];
  __shared__ float b2s[NG_];
  __shared__ float tvals[256][4];
  __shared__ int   tp[256];
  const int t = threadIdx.x;
  for (int e = t; e < NG_ * CIN; e += 256) w2s[e] = w2[e];
  if (t < NG_) b2s[t] = b2[t];
  __syncthreads();
  const int gp = blockIdx.x * 256 + t;
  const int b = gp / PIXIN, pix = gp % PIXIN;
  const int y = pix / WIN, x = pix % WIN;
  float hv[CIN];
#pragma unroll
  for (int c = 0; c < CIN; ++c) hv[c] = hin[((b * CIN + c) * HIN + y) * WIN + x];
  float m = -1e30f;
  for (int g = 0; g < NG_; ++g) {
    float l = b2s[g];
#pragma unroll
    for (int c = 0; c < CIN; ++c) l += w2s[g * CIN + c] * hv[c];
    m = fmaxf(m, l);
  }
  float es = 0.f, a0 = 0.f, a1 = 0.f, a2 = 0.f, a3 = 0.f;
  for (int g = 0; g < NG_; ++g) {
    float l = b2s[g];
#pragma unroll
    for (int c = 0; c < CIN; ++c) l += w2s[g * CIN + c] * hv[c];
    float e = __expf(l - m);
    es += e;
    a0 += e * sigx[g]; a1 += e * sigy[g]; a2 += e * opac[g]; a3 += e * rho_[g];
  }
  float inv = 1.f / es;
  tvals[t][0] = a0 * inv; tvals[t][1] = a1 * inv;
  tvals[t][2] = a2 * inv; tvals[t][3] = a3 * inv;
  tp[t] = (y % 7) * 7 + (x % 7);
  __syncthreads();
  if (t < 4 * PP) {
    int comp = t / PP, bin = t % PP;
    float s = 0.f;
    for (int tt = 0; tt < 256; ++tt)
      if (tp[tt] == bin) s += tvals[tt][comp];
    partial[blockIdx.x * (4 * PP) + t] = s;
  }
}

// ------------------------------- build the 49 resampled 14x14 splat kernels + opacity weights
__global__ void k_build_kt(const float* __restrict__ partial,
                           float* __restrict__ kt, float* __restrict__ wopn) {
  __shared__ float binsum[4 * PP];
  __shared__ float kpad[PP][14][14];
  const int t = threadIdx.x;
  if (t < 4 * PP) {
    float s = 0.f;
    for (int blk = 0; blk < 784; ++blk) s += partial[blk * (4 * PP) + t];
    binsum[t] = s;
  }
  for (int e = t; e < PP * 196; e += 256) ((float*)kpad)[e] = 0.f;
  __syncthreads();
  if (t < PP) {
    const int p = t;
    const float scl = 1.f / 1024.f;
    float wsx = binsum[0 * PP + p] * scl;
    float wsy = binsum[1 * PP + p] * scl;
    float wop = binsum[2 * PP + p] * scl;
    float wr  = binsum[3 * PP + p] * scl;
    float c00 = wsx * wsx + 1e-5f, c11 = wsy * wsy + 1e-5f;
    float c01 = wr * wsx * wsy;
    float det = c00 * c11 - c01 * c01;
    float i00 = c11 / det, i11 = c00 / det, i01 = -c01 / det;
    float nrm = 1.f / (2.f * 3.14159265358979f * sqrtf(det));
    float kv[25]; float kmax = 0.f;
#pragma unroll
    for (int a = 0; a < 5; ++a) {
      float yy = -5.f + 2.5f * a;
#pragma unroll
      for (int bx = 0; bx < 5; ++bx) {
        float xx = -5.f + 2.5f * bx;
        float z = -0.5f * (i00 * xx * xx + 2.f * i01 * xx * yy + i11 * yy * yy);
        float k = __expf(z) * nrm;
        kv[a * 5 + bx] = k;
        kmax = fmaxf(kmax, k);
      }
    }
    float ikm = 1.f / kmax;
#pragma unroll
    for (int a = 0; a < 5; ++a)
#pragma unroll
      for (int bx = 0; bx < 5; ++bx)
        kpad[p][4 + a][4 + bx] = kv[a * 5 + bx] * ikm;
    wopn[p] = wop;
  }
  __syncthreads();
  for (int e = t; e < PP * 196; e += 256) {
    int p = e / 196, r2 = e % 196, oy = r2 / 14, ox = r2 % 14;
    float tx = (0.5f - (float)(p / 7) / 7.f) * 2.f;
    float ty = (0.5f - (float)(p % 7) / 7.f) * 2.f;
    float gxv = -1.f + 2.f * ox / 13.f + tx;
    float gyv = -1.f + 2.f * oy / 13.f + ty;
    float px = (gxv + 1.f) * 0.5f * 13.f;
    float py = (gyv + 1.f) * 0.5f * 13.f;
    int x0 = (int)floorf(px), y0i = (int)floorf(py);
    float wx = px - x0, wy = py - y0i;
    float v = 0.f;
#pragma unroll
    for (int dy = 0; dy < 2; ++dy)
#pragma unroll
      for (int dx = 0; dx < 2; ++dx) {
        int yi = y0i + dy, xi = x0 + dx;
        float wv = (dy ? wy : 1.f - wy) * (dx ? wx : 1.f - wx);
        if (yi >= 0 && yi < 14 && xi >= 0 && xi < 14) v += kpad[p][yi][xi] * wv;
      }
    kt[e] = v;
  }
}

// ------------------------------------------------ per-unit splat: (64x49)@(49x196), out bf16
__global__ void k_splat(const float* __restrict__ feat, const float* __restrict__ kt,
                        const float* __restrict__ wopn, bf16_t* __restrict__ gfeat) {
  __shared__ float kts[PP * 196];
  __shared__ float cols[PP * CIN];
  const int t = threadIdx.x;
  const int u = blockIdx.x, b = blockIdx.y;
  const int r = u >> 4, c = u & 15;
  for (int e = t; e < PP * 196; e += 256) kts[e] = kt[e];
  for (int e = t; e < PP * CIN; e += 256) {
    int p = e / CIN, ch = e % CIN, i = p / 7, j = p % 7;
    cols[e] = feat[((b * CIN + ch) * HIN + r * 7 + i) * WIN + c * 7 + j] * wopn[p];
  }
  __syncthreads();
  for (int e = t; e < CIN * 196; e += 256) {
    int ch = e / 196, pix = e % 196;
    float acc = 0.f;
#pragma unroll 7
    for (int p = 0; p < PP; ++p) acc += cols[p * CIN + ch] * kts[p * 196 + pix];
    acc = fminf(fmaxf(acc, 0.f), 1.f);
    int hh = pix / 14, ww = pix % 14;
    gfeat[((b * CIN + ch) * HO + r * 14 + hh) * WO + c * 14 + ww] = f2bf(acc);
  }
}

// -------------------------------------------------- f32->bf16 weight repack (coef|freq fused)
__global__ void k_cvt(const float* __restrict__ coefw, const float* __restrict__ freqw,
                      const float* __restrict__ w1, const float* __restrict__ w2,
                      bf16_t* __restrict__ wc, bf16_t* __restrict__ w1b,
                      bf16_t* __restrict__ w2b) {
  const int idx = blockIdx.x * 256 + threadIdx.x;
  const int nwc = ROWS2 * KCONV;   // 294912
  const int nw  = HIDN * HIDN;     // 65536
  if (idx < nwc) {
    wc[idx] = f2bf(idx < nwc / 2 ? coefw[idx] : freqw[idx - nwc / 2]);
  } else if (idx < nwc + nw) {
    int j = idx - nwc; w1b[j] = f2bf(w1[j]);
  } else if (idx < nwc + 2 * nw) {
    int j = idx - nwc - nw; w2b[j] = f2bf(w2[j]);
  }
}

// ------------------ fused coef+freq 3x3 conv as WMMA GEMM: [512x576] x im2col[576x16pix]
__global__ void __launch_bounds__(256) k_conv2(
    const bf16_t* __restrict__ gfeat, const bf16_t* __restrict__ wc,
    const float* __restrict__ coefb, const float* __restrict__ freqb,
    bf16_t* __restrict__ cf) {
  __shared__ bf16_t Bs[16 * 584];   // 16 pixels x 576 K (+8 pad)
  const int t = threadIdx.x;
  const int nt = blockIdx.x, mg = blockIdx.y, b = blockIdx.z;
  const int n0 = nt * 16;
  for (int e = t; e < 16 * KCONV; e += 256) {   // cooperative im2col into LDS
    int n = e / KCONV, k = e % KCONV;
    int pix = n0 + n, y = pix / WO, x = pix % WO;
    int ci = k / 9, j = k % 9;
    int sy = y + j / 3 - 1, sx = x + j % 3 - 1;
    bf16_t v = (bf16_t)0.f;
    if (sy >= 0 && sy < HO && sx >= 0 && sx < WO)
      v = gfeat[((b * CIN + ci) * HO + sy) * WO + sx];
    Bs[n * 584 + k] = v;
  }
  __syncthreads();
  const int wave = t >> 5, lane = t & 31;
  const int rowbase = (mg * 8 + wave) * 16;
  const int lrow  = rowbase + (lane & 15);
  const int khalf = (lane >> 4) * 8;          // A: lanes<16 K{0..7,16..23}; >=16 K{8..15,24..31}
  const int bcol  = lane & 15;                // B: lanes<16 K{0..7,16..23}; >=16 K{8..15,24..31}
  const int bkoff = (lane >> 4) * 8;
  v8f acc = {};
  for (int kb = 0; kb < KCONV; kb += 32) {
    union { v16bf v; v8bf h[2]; } ua, ub;
    const bf16_t* ap = wc + lrow * KCONV + kb + khalf;
    ua.h[0] = *(const v8bf*)ap;
    ua.h[1] = *(const v8bf*)(ap + 16);
    const bf16_t* bp = &Bs[bcol * 584 + kb + bkoff];
    ub.h[0] = *(const v8bf*)bp;
    ub.h[1] = *(const v8bf*)(bp + 16);
    if (kb + 32 < KCONV) __builtin_prefetch(ap + 32, 0, 0);
    acc = __builtin_amdgcn_wmma_f32_16x16x32_bf16(false, ua.v, false, ub.v,
                                                  (short)0, acc, false, false);
  }
  const int npix = n0 + bcol;
  const int y = npix / WO, x = npix % WO;
  const int m0 = (lane >> 4) * 8;
#pragma unroll
  for (int rI = 0; rI < 8; ++rI) {
    int row = rowbase + m0 + rI;
    float v = acc[rI] + (row < HIDN ? coefb[row] : freqb[row - HIDN]);
    cf[((b * HO + y) * WO + x) * ROWS2 + row] = f2bf(v);   // pixel-major for gather
  }
}

// ------------------------- gather + sin/cos basis -> MLP input rows X[q][256] (wave per query)
__global__ void k_query(const float* __restrict__ coord, const float* __restrict__ cell,
                        const float* __restrict__ phasew, const bf16_t* __restrict__ cf,
                        bf16_t* __restrict__ X) {
  const int t = threadIdx.x, wave = t >> 5, lane = t & 31;
  const int gq = blockIdx.x * 8 + wave;
  const int b = gq / QPI;
  const float gy = coord[gq * 2 + 0], gx = coord[gq * 2 + 1];
  const float cl0 = cell[gq * 2 + 0], cl1 = cell[gq * 2 + 1];
  const int ix = (int)rintf(((gx + 1.f) * WO - 1.f) * 0.5f);
  const int iy = (int)rintf(((gy + 1.f) * HO - 1.f) * 0.5f);
  const bool ok = (ix >= 0) & (ix < WO) & (iy >= 0) & (iy < HO);
  const int ixc = min(max(ix, 0), WO - 1);
  const int iyc = min(max(iy, 0), HO - 1);
  const float qcy = ok ? ((2.f * iyc + 1.f) / HO - 1.f) : 0.f;
  const float qcx = ok ? ((2.f * ixc + 1.f) / WO - 1.f) : 0.f;
  const float rel0 = (gy - qcy) * HO, rel1 = (gx - qcx) * WO;
  const float cr0 = cl0 * HO, cr1 = cl1 * WO;
  const int base = ((b * HO + iyc) * WO + ixc) * ROWS2;
#pragma unroll
  for (int i = 0; i < 4; ++i) {
    const int k = lane + 32 * i;
    float f0 = 0.f, f1 = 0.f, c0 = 0.f, c1 = 0.f;
    if (ok) {
      f0 = (float)cf[base + HIDN + 2 * k];
      f1 = (float)cf[base + HIDN + 2 * k + 1];
      c0 = (float)cf[base + k];
      c1 = (float)cf[base + 128 + k];
    }
    const float ph = phasew[2 * k] * cr0 + phasew[2 * k + 1] * cr1;
    const float ang = 3.14159265358979f * (f0 * rel0 + f1 * rel1 + ph);
    X[gq * HIDN + k]       = f2bf(c0 * __cosf(ang));
    X[gq * HIDN + 128 + k] = f2bf(c1 * __sinf(ang));
  }
}

// --------------------------------------------- WMMA MLP layer: out = act(in @ W^T + b), bf16
__global__ void __launch_bounds__(256) k_mlp(
    const bf16_t* __restrict__ in, const bf16_t* __restrict__ w,
    const float* __restrict__ bias, bf16_t* __restrict__ out, const int do_relu) {
  __shared__ bf16_t Xs[16 * 264];
  const int t = threadIdx.x;
  const int qb = blockIdx.x * 16;
  for (int e = t; e < 16 * HIDN; e += 256) {
    int n = e / HIDN, k = e % HIDN;
    Xs[n * 264 + k] = in[(qb + n) * HIDN + k];
  }
  __syncthreads();
  const int wave = t >> 5, lane = t & 31;
  const int arow = lane & 15;
  const int akhalf = (lane >> 4) * 8;
  const int bkoff = (lane >> 4) * 8;
#pragma unroll
  for (int half = 0; half < 2; ++half) {
    const int nb = (wave + half * 8) * 16;
    const int brow = nb + (lane & 15);
    v8f acc = {};
#pragma unroll
    for (int kb = 0; kb < HIDN; kb += 32) {
      union { v16bf v; v8bf h[2]; } ua, ub;
      const bf16_t* ap = &Xs[arow * 264 + kb + akhalf];
      ua.h[0] = *(const v8bf*)ap;
      ua.h[1] = *(const v8bf*)(ap + 16);
      const bf16_t* bp = w + brow * HIDN + kb + bkoff;
      ub.h[0] = *(const v8bf*)bp;
      ub.h[1] = *(const v8bf*)(bp + 16);
      acc = __builtin_amdgcn_wmma_f32_16x16x32_bf16(false, ua.v, false, ub.v,
                                                    (short)0, acc, false, false);
    }
    const int n = nb + (lane & 15);
    const int m0 = (lane >> 4) * 8;
    const float bs = bias[n];
#pragma unroll
    for (int rI = 0; rI < 8; ++rI) {
      float v = acc[rI] + bs;
      if (do_relu) v = fmaxf(v, 0.f);
      out[(qb + m0 + rI) * HIDN + n] = f2bf(v);
    }
  }
}

// ---------------------------------------------------------------- final 256->3 head (f32)
__global__ void k_mlp3(const bf16_t* __restrict__ y2, const float* __restrict__ w3,
                       const float* __restrict__ b3, float* __restrict__ outp) {
  const int gq = blockIdx.x * 256 + threadIdx.x;
  float a0 = b3[0], a1 = b3[1], a2 = b3[2];
  const bf16_t* row = y2 + gq * HIDN;
  for (int k = 0; k < HIDN; ++k) {
    float v = (float)row[k];
    a0 += v * w3[k];
    a1 += v * w3[HIDN + k];
    a2 += v * w3[2 * HIDN + k];
  }
  outp[gq * 3 + 0] = a0; outp[gq * 3 + 1] = a1; outp[gq * 3 + 2] = a2;
}

extern "C" void kernel_launch(void* const* d_in, const int* in_sizes, int n_in,
                              void* d_out, int out_size, void* d_ws, size_t ws_size,
                              hipStream_t stream) {
  (void)in_sizes; (void)n_in; (void)out_size; (void)ws_size;
  const float* feat   = (const float*)d_in[0];
  const float* coord  = (const float*)d_in[1];
  const float* cell   = (const float*)d_in[2];
  const float* cls_w1 = (const float*)d_in[3];
  const float* cls_b1 = (const float*)d_in[4];
  const float* cls_w2 = (const float*)d_in[5];
  const float* cls_b2 = (const float*)d_in[6];
  const float* sigx   = (const float*)d_in[7];
  const float* sigy   = (const float*)d_in[8];
  const float* opac   = (const float*)d_in[9];
  const float* rho_   = (const float*)d_in[10];
  const float* coefw  = (const float*)d_in[11];
  const float* coefb  = (const float*)d_in[12];
  const float* freqw  = (const float*)d_in[13];
  const float* freqb  = (const float*)d_in[14];
  const float* phasew = (const float*)d_in[15];
  const float* mlpw1  = (const float*)d_in[16];
  const float* mlpb1  = (const float*)d_in[17];
  const float* mlpw2  = (const float*)d_in[18];
  const float* mlpb2  = (const float*)d_in[19];
  const float* mlpw3  = (const float*)d_in[20];
  const float* mlpb3  = (const float*)d_in[21];

  char* ws = (char*)d_ws;
  size_t off = 0;
  auto take = [&](size_t bytes) {
    char* p = ws + off;
    off = (off + bytes + 255) & ~(size_t)255;
    return p;
  };
  float*  h_f   = (float*) take((size_t)NB * CIN * HIN * WIN * 4);
  float*  part  = (float*) take((size_t)784 * 4 * PP * 4);
  float*  kt    = (float*) take((size_t)PP * 196 * 4);
  float*  wopn  = (float*) take((size_t)PP * 4);
  bf16_t* gfeat = (bf16_t*)take((size_t)NB * CIN * HO * WO * 2);
  bf16_t* wc    = (bf16_t*)take((size_t)ROWS2 * KCONV * 2);
  bf16_t* w1b   = (bf16_t*)take((size_t)HIDN * HIDN * 2);
  bf16_t* w2b   = (bf16_t*)take((size_t)HIDN * HIDN * 2);
  bf16_t* cfbuf = (bf16_t*)take((size_t)NB * HO * WO * ROWS2 * 2);
  bf16_t* Xbuf  = (bf16_t*)take((size_t)QTOT * HIDN * 2);
  bf16_t* Y1    = (bf16_t*)take((size_t)QTOT * HIDN * 2);
  bf16_t* Y2    = Xbuf;   // X dead after layer 1 -> reuse

  k_conv1<<<dim3(49, CIN, NB), 256, 0, stream>>>(feat, cls_w1, cls_b1, h_f);
  k_logits<<<784, 256, 0, stream>>>(h_f, cls_w2, cls_b2, sigx, sigy, opac, rho_, part);
  k_build_kt<<<1, 256, 0, stream>>>(part, kt, wopn);
  k_splat<<<dim3(256, NB), 256, 0, stream>>>(feat, kt, wopn, gfeat);
  {
    const int total = ROWS2 * KCONV + 2 * HIDN * HIDN;
    k_cvt<<<(total + 255) / 256, 256, 0, stream>>>(coefw, freqw, mlpw1, mlpw2, wc, w1b, w2b);
  }
  k_conv2<<<dim3(PIXO / 16, 4, NB), 256, 0, stream>>>(gfeat, wc, coefb, freqb, cfbuf);
  k_query<<<QTOT / 8, 256, 0, stream>>>(coord, cell, phasew, cfbuf, Xbuf);
  k_mlp<<<QTOT / 16, 256, 0, stream>>>(Xbuf, w1b, mlpb1, Y1, 1);
  k_mlp<<<QTOT / 16, 256, 0, stream>>>(Y1, w2b, mlpb2, Y2, 1);
  k_mlp3<<<QTOT / 256, 256, 0, stream>>>(Y2, mlpw3, mlpb3, (float*)d_out);
}
